// AttentionLayer_37598143709275
// MI455X (gfx1250) — compile-verified, matrix operand order checked by reading, not run on hardware
//
#include <hip/hip_runtime.h>

// ---------------------------------------------------------------------------
// AttentionLayer for MI455X (gfx1250): bf16 WMMA GEMMs + flash attention.
// B=4, S=2048, E=1024, heads=16, head_dim=64 (RoPE only), SDPA over full E.
// GEMMs use 32x64 wave tiles (8 WMMAs / 12 b128 loads per K-step), software
// pipelined with explicit ping-pong fragment sets (no register rotation).
// ---------------------------------------------------------------------------

#define B_  4
#define S_  2048
#define E_  1024
#define BS  (B_ * S_)          // 8192 rows
#define LN10000 9.210340371976184f

typedef __attribute__((ext_vector_type(16))) __bf16 v16bf;
typedef __attribute__((ext_vector_type(8)))  __bf16 v8bf;
typedef __attribute__((ext_vector_type(8)))  float  v8f;
typedef __attribute__((ext_vector_type(8)))  unsigned short v8u16;

// ---- bf16 bit helpers (cold/elementwise paths only) ------------------------
__device__ __forceinline__ unsigned short f2bf_bits(float f) {
  union { float f; unsigned u; } x{f};
  unsigned r = x.u + 0x7FFFu + ((x.u >> 16) & 1u);   // round-nearest-even
  return (unsigned short)(r >> 16);
}
__device__ __forceinline__ __bf16 bits2bf(unsigned short h) {
  union { unsigned short u; __bf16 b; } x; x.u = h; return x.b;
}
__device__ __forceinline__ float bf2f(unsigned short h) {
  union { unsigned u; float f; } x; x.u = ((unsigned)h) << 16; return x.f;
}

// ---- WMMA wrapper: D = A(16x32 bf16) * B(32x16 bf16) + C(16x16 f32) --------
__device__ __forceinline__ v8f wmma_bf16(v16bf a, v16bf b, v8f c) {
  return __builtin_amdgcn_wmma_f32_16x16x32_bf16(
      /*neg_a=*/false, a, /*neg_b=*/false, b,
      /*c_mod=*/(short)0, c, /*reuse_a=*/false, /*reuse_b=*/false);
}

// K offset held by vector element pair (2j,2j+1) for lane-half `hi`
// (ISA 7.12.2: lanes 0-15 hold K {0..7,16..23}, lanes 16-31 hold {8..15,24..31})
__device__ __forceinline__ int kpat(int j, int hi) {
  return ((j & 4) ? 16 : 0) + hi * 8 + (j & 3) * 2;
}

// Load one A/B fragment's 32-K slice for this lane from a bf16 row pointer.
// Lane's 16 values are two contiguous 16B runs -> two global_load_b128.
__device__ __forceinline__ v16bf load_k32(const unsigned short* rowk0, int hi) {
  const v8bf lo = *(const v8bf*)(rowk0 + hi * 8);        // K {0..7}/{8..15}
  const v8bf hh = *(const v8bf*)(rowk0 + 16 + hi * 8);   // K {16..23}/{24..31}
  return __builtin_shufflevector(lo, hh, 0, 1, 2, 3, 4, 5, 6, 7,
                                 8, 9, 10, 11, 12, 13, 14, 15);
}

// ---------------------------------------------------------------------------
// Kernel 0: f32 -> bf16 bulk convert (8 elements per thread).
// ---------------------------------------------------------------------------
__global__ __launch_bounds__(256) void cvt_bf16_kernel(
    const float* __restrict__ src, unsigned short* __restrict__ dst) {
  size_t i = (size_t)blockIdx.x * blockDim.x + threadIdx.x;
  v8f x = ((const v8f*)src)[i];
  v8u16 o;
#pragma unroll
  for (int j = 0; j < 8; ++j) o[j] = f2bf_bits(x[j]);
  ((v8u16*)dst)[i] = o;
}

// ---------------------------------------------------------------------------
// Software-pipelined 32x64 wave-tile GEMM body (ping-pong fragment sets).
// A: rows [m0, m0+32) of Amat (row-major, K = E_);  B: rows [n0, n0+64).
// Accumulators c[t*2+r] hold rows r*16.., cols t*16.. of the tile.
// ---------------------------------------------------------------------------
__device__ __forceinline__ void load_set(
    const unsigned short* __restrict__ arow0, const unsigned short* __restrict__ arow1,
    const unsigned short* __restrict__ wrow0, int k, int hi,
    v16bf& a0, v16bf& a1, v16bf& b0, v16bf& b1, v16bf& b2, v16bf& b3) {
  a0 = load_k32(arow0 + k, hi);
  a1 = load_k32(arow1 + k, hi);
  b0 = load_k32(wrow0 + k, hi);
  b1 = load_k32(wrow0 + (size_t)16 * E_ + k, hi);
  b2 = load_k32(wrow0 + (size_t)32 * E_ + k, hi);
  b3 = load_k32(wrow0 + (size_t)48 * E_ + k, hi);
}

__device__ __forceinline__ void wmma_step(v8f c[8], v16bf a0, v16bf a1,
                                          v16bf b0, v16bf b1, v16bf b2, v16bf b3) {
  c[0] = wmma_bf16(a0, b0, c[0]);  c[1] = wmma_bf16(a1, b0, c[1]);
  c[2] = wmma_bf16(a0, b1, c[2]);  c[3] = wmma_bf16(a1, b1, c[3]);
  c[4] = wmma_bf16(a0, b2, c[4]);  c[5] = wmma_bf16(a1, b2, c[5]);
  c[6] = wmma_bf16(a0, b3, c[6]);  c[7] = wmma_bf16(a1, b3, c[7]);
}

__device__ __forceinline__ void gemm_tile_32x64(
    const unsigned short* __restrict__ arow0, const unsigned short* __restrict__ wrow0,
    int hi, v8f c[8]) {
  const unsigned short* arow1 = arow0 + (size_t)16 * E_;
  v16bf A0, A1, B0, B1, B2, B3;        // ping set
  v16bf C0, C1, D0, D1, D2, D3;        // pong set
  load_set(arow0, arow1, wrow0, 0, hi, A0, A1, B0, B1, B2, B3);
#pragma unroll 1
  for (int k0 = 0; k0 < E_ - 64; k0 += 64) {
    load_set(arow0, arow1, wrow0, k0 + 32, hi, C0, C1, D0, D1, D2, D3);
    wmma_step(c, A0, A1, B0, B1, B2, B3);
    load_set(arow0, arow1, wrow0, k0 + 64, hi, A0, A1, B0, B1, B2, B3);
    wmma_step(c, C0, C1, D0, D1, D2, D3);
  }
  load_set(arow0, arow1, wrow0, E_ - 32, hi, C0, C1, D0, D1, D2, D3);
  wmma_step(c, A0, A1, B0, B1, B2, B3);
  wmma_step(c, C0, C1, D0, D1, D2, D3);
}

// ---------------------------------------------------------------------------
// Kernel 1: qkv = X (BSxE bf16) * Wqkv^T (3E x E bf16) -> q,k,v bf16 buffers.
// ---------------------------------------------------------------------------
__global__ __launch_bounds__(256) void qkv_gemm_kernel(
    const unsigned short* __restrict__ X, const unsigned short* __restrict__ W,
    unsigned short* __restrict__ qb, unsigned short* __restrict__ kb,
    unsigned short* __restrict__ vb) {
  const int lane = threadIdx.x & 31;
  const int wave = threadIdx.x >> 5;
  const int TN = (3 * E_) / 64;                 // 48 col-tiles
  const int tile = blockIdx.x * 8 + wave;       // 256*48 = 12288 tiles
  const int tm = tile / TN, tn = tile % TN;
  const int m  = lane & 15, hi = lane >> 4;

  const unsigned short* arow0 = X + (size_t)(tm * 32 + m) * E_;
  const unsigned short* wrow0 = W + (size_t)(tn * 64 + m) * E_;
  v8f c[8] = {v8f{}, v8f{}, v8f{}, v8f{}, v8f{}, v8f{}, v8f{}, v8f{}};

  gemm_tile_32x64(arow0, wrow0, hi, c);

  // C layout: VGPR i -> rows i (lanes 0-15) / i+8 (lanes 16-31); col = lane&15
#pragma unroll
  for (int t = 0; t < 4; ++t) {
#pragma unroll
    for (int r = 0; r < 2; ++r) {
#pragma unroll
      for (int i = 0; i < 8; ++i) {
        int row = tm * 32 + r * 16 + i + 8 * hi;
        int col = tn * 64 + t * 16 + m;
        unsigned short hv = f2bf_bits(c[t * 2 + r][i]);
        if      (col < E_)     qb[(size_t)row * E_ + col]            = hv;
        else if (col < 2 * E_) kb[(size_t)row * E_ + (col - E_)]     = hv;
        else                   vb[(size_t)row * E_ + (col - 2 * E_)] = hv;
      }
    }
  }
}

// ---------------------------------------------------------------------------
// Kernel 2: interleaved-pair RoPE, in place on bf16 q and k.
// ---------------------------------------------------------------------------
__global__ __launch_bounds__(256) void rope_kernel(
    unsigned short* __restrict__ qb, unsigned short* __restrict__ kb) {
  const size_t pairs = (size_t)BS * (E_ / 2);   // per tensor
  size_t gid = (size_t)blockIdx.x * blockDim.x + threadIdx.x;
  unsigned short* buf = (gid < pairs) ? qb : kb;
  size_t p = (gid < pairs) ? gid : gid - pairs;
  int    pe  = (int)(p % (E_ / 2));
  size_t row = p / (E_ / 2);                    // b*S + s
  int s  = (int)(row % S_);
  int h  = pe >> 5;                             // head (64-dim -> 32 pairs)
  int pp = pe & 31;                             // pair within head
  int e0 = h * 64 + pp * 2;
  float inv_freq = __expf(-((float)(2 * pp) / 64.0f) * LN10000);
  float ang = (float)s * inv_freq;
  float cs = __cosf(ang), sn = __sinf(ang);
  size_t idx = row * E_ + e0;
  float x0 = bf2f(buf[idx]), x1 = bf2f(buf[idx + 1]);
  buf[idx]     = f2bf_bits(x0 * cs - x1 * sn);
  buf[idx + 1] = f2bf_bits(x0 * sn + x1 * cs);
}

// ---------------------------------------------------------------------------
// Kernel 3: causal flash attention (single logical head over E=1024).
// 1 block = 1 (batch, 16-query tile); 16 waves; wave w owns out cols [64w,64w+64).
// Scores for each 32-key block are cooperatively reduced into LDS (ds_add_f32),
// each wave then runs replicated online softmax and a WMMA P*V update.
// ---------------------------------------------------------------------------
__global__ __launch_bounds__(512) void flash_attn_kernel(
    const unsigned short* __restrict__ qb, const unsigned short* __restrict__ kb,
    const unsigned short* __restrict__ vb, unsigned short* __restrict__ ao) {
  __shared__ float smem[16 * 32];
  const int lane = threadIdx.x & 31;
  const int wave = threadIdx.x >> 5;            // 0..15
  const int b  = blockIdx.x >> 7;               // 128 q-tiles per batch
  const int qt = blockIdx.x & 127;
  const int q0 = qt * 16;
  const int m  = lane & 15, hi = lane >> 4;
  const int row_g = b * S_ + q0;
  const float NEG = -__builtin_inff();
  const float scale = 0.125f;                   // 64^-0.5

  const unsigned short* qrow  = qb + (size_t)(row_g + m) * E_;
  v8f c[4] = {v8f{}, v8f{}, v8f{}, v8f{}};
  float rm = NEG, rs = 0.0f;

  const int nkb = (q0 + 16 + 31) / 32;          // 32-key blocks (causal bound)
  for (int kbk = 0; kbk < nkb; ++kbk) {
    const int kstart = kbk * 32;

    smem[threadIdx.x] = 0.0f;                   // 512 threads == 512 floats
    __syncthreads();

    // --- cooperative partial scores: this wave handles E-chunks 2w, 2w+1 ---
    v8f s0 = {}, s1 = {};
    const unsigned short* krow0 = kb + (size_t)(b * S_ + kstart + m) * E_;
    const unsigned short* krow1 = krow0 + (size_t)16 * E_;
#pragma unroll
    for (int cc = 0; cc < 2; ++cc) {
      const int e0 = (2 * wave + cc) * 32;
      v16bf aq = load_k32(qrow  + e0, hi);
      v16bf b0 = load_k32(krow0 + e0, hi);
      v16bf b1 = load_k32(krow1 + e0, hi);
      s0 = wmma_bf16(aq, b0, s0);               // keys kstart..+15
      s1 = wmma_bf16(aq, b1, s1);               // keys kstart+16..+31
    }
#pragma unroll
    for (int i = 0; i < 8; ++i) {
      int r = i + 8 * hi;
      atomicAdd(&smem[r * 32 + m],      s0[i]); // ds_add_f32
      atomicAdd(&smem[r * 32 + m + 16], s1[i]);
    }
    __syncthreads();

    // --- replicated online softmax; read P in A-matrix layout (row = m) ---
    float pv[16];
    float bmax = NEG;
#pragma unroll
    for (int j = 0; j < 8; ++j) {
      int kk = kpat(j, hi);
#pragma unroll
      for (int u = 0; u < 2; ++u) {
        int kloc = kk + u;
        float v = (kstart + kloc <= q0 + m) ? smem[m * 32 + kloc] * scale : NEG;
        pv[2 * j + u] = v;
        bmax = fmaxf(bmax, v);
      }
    }
    bmax = fmaxf(bmax, __shfl_xor(bmax, 16));   // combine K halves of row m
    float nm = fmaxf(rm, bmax);
    float lsum = 0.0f;
    v16bf ap;
#pragma unroll
    for (int j = 0; j < 16; ++j) {
      float p = __expf(pv[j] - nm);             // exp(-inf)=0 for masked
      lsum += p;
      ap[j] = bits2bf(f2bf_bits(p));
    }
    float bs = lsum + __shfl_xor(lsum, 16);
    float alpha = __expf(rm - nm);
    rs = rs * alpha + bs;
    rm = nm;

    // rescale accumulators (row r stats live at lane r, r in 0..15)
#pragma unroll
    for (int i = 0; i < 8; ++i) {
      float alr = __shfl(alpha, i + 8 * hi);
#pragma unroll
      for (int t = 0; t < 4; ++t) c[t][i] *= alr;
    }

    // --- P * V for this wave's 64 output columns ---
#pragma unroll
    for (int t = 0; t < 4; ++t) {
      int e = wave * 64 + t * 16 + m;
      v16bf bvv;
#pragma unroll
      for (int j = 0; j < 8; ++j) {
        int kk = kpat(j, hi);
        bvv[2 * j]     = bits2bf(vb[(size_t)(b * S_ + kstart + kk)     * E_ + e]);
        bvv[2 * j + 1] = bits2bf(vb[(size_t)(b * S_ + kstart + kk + 1) * E_ + e]);
      }
      c[t] = wmma_bf16(ap, bvv, c[t]);
    }
    __syncthreads();                            // smem reused next key block
  }

  // normalize and store bf16
#pragma unroll
  for (int i = 0; i < 8; ++i) {
    float sr  = __shfl(rs, i + 8 * hi);
    float inv = 1.0f / sr;
    int row = row_g + i + 8 * hi;
#pragma unroll
    for (int t = 0; t < 4; ++t)
      ao[(size_t)row * E_ + wave * 64 + t * 16 + m] = f2bf_bits(c[t][i] * inv);
  }
}

// ---------------------------------------------------------------------------
// Kernel 4: final projection out = attn (BSxE bf16) * Wout^T (E x E bf16) -> f32
// ---------------------------------------------------------------------------
__global__ __launch_bounds__(256) void out_gemm_kernel(
    const unsigned short* __restrict__ A, const unsigned short* __restrict__ W,
    float* __restrict__ out) {
  const int lane = threadIdx.x & 31;
  const int wave = threadIdx.x >> 5;
  const int TN = E_ / 64;                       // 16 col-tiles
  const int tile = blockIdx.x * 8 + wave;       // 256*16 = 4096 tiles
  const int tm = tile / TN, tn = tile % TN;
  const int m  = lane & 15, hi = lane >> 4;

  const unsigned short* arow0 = A + (size_t)(tm * 32 + m) * E_;
  const unsigned short* wrow0 = W + (size_t)(tn * 64 + m) * E_;
  v8f c[8] = {v8f{}, v8f{}, v8f{}, v8f{}, v8f{}, v8f{}, v8f{}, v8f{}};

  gemm_tile_32x64(arow0, wrow0, hi, c);

#pragma unroll
  for (int t = 0; t < 4; ++t)
#pragma unroll
    for (int r = 0; r < 2; ++r)
#pragma unroll
      for (int i = 0; i < 8; ++i)
        out[(size_t)(tm * 32 + r * 16 + i + 8 * hi) * E_ + tn * 64 + t * 16 + m]
            = c[t * 2 + r][i];
}

// ---------------------------------------------------------------------------
extern "C" void kernel_launch(void* const* d_in, const int* in_sizes, int n_in,
                              void* d_out, int out_size, void* d_ws, size_t ws_size,
                              hipStream_t stream) {
  const float* keyvalue = (const float*)d_in[0];
  // d_in[1] = query (unused by the self-attention reference path)
  const float* Wqkv = (const float*)d_in[2];
  const float* Wout = (const float*)d_in[3];
  float* out = (float*)d_out;

  // workspace layout (bf16 = u16):
  //   Xb (8M) | Wqkvb (3M) | Woutb (1M) | q (8M) | k (8M) | v (8M) | attn (8M)
  unsigned short* Xb  = (unsigned short*)d_ws;
  unsigned short* Wqb = Xb  + (size_t)BS * E_;
  unsigned short* Wob = Wqb + (size_t)3 * E_ * E_;
  unsigned short* qb  = Wob + (size_t)E_ * E_;
  unsigned short* kb  = qb  + (size_t)BS * E_;
  unsigned short* vb  = kb  + (size_t)BS * E_;
  unsigned short* ao  = vb  + (size_t)BS * E_;   // total 88 MiB

  // 0) one-shot f32->bf16 conversions (8 elems/thread)
  cvt_bf16_kernel<<<(BS * E_) / 8 / 256, 256, 0, stream>>>(keyvalue, Xb);
  cvt_bf16_kernel<<<(3 * E_ * E_) / 8 / 256, 256, 0, stream>>>(Wqkv, Wqb);
  cvt_bf16_kernel<<<(E_ * E_) / 8 / 256, 256, 0, stream>>>(Wout, Wob);
  // 1) QKV GEMM: 256*48 = 12288 wave-tiles / 8 waves per block
  qkv_gemm_kernel<<<1536, 256, 0, stream>>>(Xb, Wqb, qb, kb, vb);
  // 2) RoPE on q and k: 2 * BS * E/2 = 8388608 threads
  rope_kernel<<<32768, 256, 0, stream>>>(qb, kb);
  // 3) flash attention: one block per (batch, 16-query tile)
  flash_attn_kernel<<<B_ * (S_ / 16), 512, 0, stream>>>(qb, kb, vb, ao);
  // 4) output projection: 256*16 = 4096 wave-tiles / 8 waves per block
  out_gemm_kernel<<<512, 256, 0, stream>>>(ao, Wob, out);
}
